// CustomDotProductAttention_78331613544602
// MI455X (gfx1250) — matrix operand order repaired and checked by reading, not compiled, not run on hardware
//
#include <hip/hip_runtime.h>
#include <stdint.h>

typedef __attribute__((ext_vector_type(16))) __bf16   v16bf;
typedef __attribute__((ext_vector_type(8)))  __bf16   v8bf;
typedef __attribute__((ext_vector_type(8)))  float    v8f;
typedef __attribute__((ext_vector_type(4)))  float    v4f;
typedef __attribute__((ext_vector_type(4)))  uint32_t u32x4;
typedef __attribute__((ext_vector_type(8)))  uint32_t u32x8;

#define S_LEN 2048
#define BATCH 4
#define HEADS 16
#define DHEAD 128
#define ROWSTRIDE (BATCH * HEADS * DHEAD)   // 8192 floats between seq positions

__device__ __forceinline__ float half16_max(float x) {
    x = fmaxf(x, __shfl_xor(x, 1, 32));
    x = fmaxf(x, __shfl_xor(x, 2, 32));
    x = fmaxf(x, __shfl_xor(x, 4, 32));
    x = fmaxf(x, __shfl_xor(x, 8, 32));
    return x;
}
__device__ __forceinline__ float half16_sum(float x) {
    x += __shfl_xor(x, 1, 32);
    x += __shfl_xor(x, 2, 32);
    x += __shfl_xor(x, 4, 32);
    x += __shfl_xor(x, 8, 32);
    return x;
}

// Issue a TDM load of a 32-row x 128-col f32 tile (row stride ROWSTRIDE floats)
// from global memory into LDS at byte offset lds_off. 2D tensor -> 2 D# groups.
__device__ __forceinline__ void tdm_load_tile32(const float* gtile, uint32_t lds_off) {
    uint64_t ga = (uint64_t)(uintptr_t)gtile;
    u32x4 g0;
    g0.x = 1u;                                    // count=1, no gather
    g0.y = lds_off;                               // lds_addr (bytes)
    g0.z = (uint32_t)ga;                          // global_addr[31:0]
    g0.w = (uint32_t)(ga >> 32) | (2u << 30);     // global_addr[56:32] | type=2
    u32x8 g1;
    g1[0] = 2u << 16;                             // data_size = 4 bytes
    g1[1] = ((uint32_t)DHEAD) << 16;              // tensor_dim0 = 128 (bits 63:48)
    g1[2] = ((uint32_t)S_LEN) << 16;              // tensor_dim1 = 2048 (bits 111:96->lo16 at 95:80.. packed)
    g1[3] = ((uint32_t)DHEAD) << 16;              // tile_dim0 = 128 (bits 127:112)
    g1[4] = 32u;                                  // tile_dim1 = 32 rows, tile_dim2 = 0
    g1[5] = (uint32_t)ROWSTRIDE;                  // tensor_dim0_stride = 8192 (48-bit lo)
    g1[6] = 0u;                                   // stride hi | tensor_dim1_stride lo
    g1[7] = 0u;
    asm volatile("tensor_load_to_lds %0, %1" :: "s"(g0), "s"(g1) : "memory");
}

// Flash attention fwd, causal, fp32 in/out, bf16 WMMA compute.
// One wave = one 16-row q tile; 8 waves/block = 128 q rows. All waves run the
// block-max causal key range so workgroup barriers + shared TDM staging are legal.
__global__ __launch_bounds__(256, 1)
void fa_fwd_kernel(const float* __restrict__ Q, const float* __restrict__ K,
                   const float* __restrict__ V, float* __restrict__ O) {
    __shared__ alignas(64) float  fK[2][32][DHEAD];    // TDM staging (f32) 32 KB
    __shared__ alignas(64) float  fV[2][32][DHEAD];    // TDM staging (f32) 32 KB
    __shared__ alignas(64) __bf16 kbf[32][DHEAD];      // bf16 K tile        8 KB
    __shared__ alignas(64) __bf16 vT[DHEAD][32];       // bf16 V^T tile      8 KB
    __shared__ alignas(64) __bf16 pT[8][16][32];       // per-wave P bounce  8 KB

    const int tid  = threadIdx.x;
    const int wv   = tid >> 5;
    const int lane = tid & 31;
    const int hl   = lane >> 4;
    const int lx   = lane & 15;

    const int bh     = blockIdx.y;          // b*H + h
    const int bh_off = bh * DHEAD;
    const int q0     = blockIdx.x * 128;
    const int qbase  = q0 + wv * 16;

    const float scale_log2e = 0.08838834764831845f * 1.4426950408889634f;

    // ---- Q tile -> bf16 A fragments (4 chunks of K=32 over D=128)
    v16bf qa[4];
    {
        const float* qrow = Q + (size_t)(qbase + lx) * ROWSTRIDE + bh_off;
        #pragma unroll
        for (int c = 0; c < 4; ++c) {
            const int d0 = c * 32 + hl * 16;
            v4f f0 = *(const v4f*)(qrow + d0);
            v4f f1 = *(const v4f*)(qrow + d0 + 4);
            v4f f2 = *(const v4f*)(qrow + d0 + 8);
            v4f f3 = *(const v4f*)(qrow + d0 + 12);
            #pragma unroll
            for (int e = 0; e < 4; ++e) {
                qa[c][e]      = (__bf16)f0[e];
                qa[c][e + 4]  = (__bf16)f1[e];
                qa[c][e + 8]  = (__bf16)f2[e];
                qa[c][e + 12] = (__bf16)f3[e];
            }
        }
    }

    v8f acc[8];
    #pragma unroll
    for (int nt = 0; nt < 8; ++nt)
        #pragma unroll
        for (int v = 0; v < 8; ++v) acc[nt][v] = 0.f;

    float rm[8], rl[8];
    #pragma unroll
    for (int v = 0; v < 8; ++v) { rm[v] = -1e30f; rl[v] = 0.f; }

    const int nblk = (q0 + 128) >> 5;       // block-max causal key blocks of 32
    const float* Kbh = K + bh_off;
    const float* Vbh = V + bh_off;

    // prologue: stage key block 0 via Tensor Data Mover
    if (wv == 0) {
        tdm_load_tile32(Kbh, (uint32_t)(uintptr_t)&fK[0][0][0]);
        tdm_load_tile32(Vbh, (uint32_t)(uintptr_t)&fV[0][0][0]);
    }

    int cur = 0;
    for (int j = 0; j < nblk; ++j) {
        const int kb = j << 5;

        if (wv == 0) __builtin_amdgcn_s_wait_tensorcnt(0);
        __syncthreads();                     // f32 tiles[cur] visible

        if ((wv == 0) && (j + 1 < nblk)) {   // overlap next DMA with compute
            const size_t off = (size_t)(kb + 32) * ROWSTRIDE;
            tdm_load_tile32(Kbh + off, (uint32_t)(uintptr_t)&fK[cur ^ 1][0][0]);
            tdm_load_tile32(Vbh + off, (uint32_t)(uintptr_t)&fV[cur ^ 1][0][0]);
        }

        // ---- shared conversion pass: f32 tiles -> bf16 K (row major) + V^T
        {
            const float* fk = &fK[cur][0][0];
            __bf16* kd = &kbf[0][0];
            #pragma unroll
            for (int i = 0; i < 16; i += 4) {
                const int idx = tid * 16 + i;
                v4f f = *(const v4f*)(fk + idx);
                kd[idx]     = (__bf16)f.x;
                kd[idx + 1] = (__bf16)f.y;
                kd[idx + 2] = (__bf16)f.z;
                kd[idx + 3] = (__bf16)f.w;
            }
            const float* fv = &fV[cur][0][0];
            const int d  = tid >> 1;
            const int k0 = (tid & 1) << 4;
            #pragma unroll
            for (int e = 0; e < 16; ++e)
                vT[d][k0 + e] = (__bf16)fv[(size_t)(k0 + e) * DHEAD + d];
        }
        __syncthreads();                     // bf16 tiles ready

        if (kb <= qbase + 15) {              // this wave's causal range
            // ---- S = Q K^T (16 x 32): two 16-key halves, 4 d-chunks
            v8f sf[2];
            #pragma unroll
            for (int t = 0; t < 2; ++t) {
                #pragma unroll
                for (int v = 0; v < 8; ++v) sf[t][v] = 0.f;
                #pragma unroll
                for (int c = 0; c < 4; ++c) {
                    v16bf kfrag = *(const v16bf*)&kbf[t * 16 + lx][c * 32 + hl * 16];
                    sf[t] = __builtin_amdgcn_wmma_f32_16x16x32_bf16(
                        false, qa[c], false, kfrag, (short)0, sf[t], false, false);
                }
            }

            // ---- scale + causal mask + online softmax (C layout: M=v+8*hl, N=lx)
            float alpha[8];
            #pragma unroll
            for (int v = 0; v < 8; ++v) {
                const int qrow = qbase + v + hl * 8;
                float s0 = ((kb + lx)      <= qrow) ? sf[0][v] * scale_log2e : -1e30f;
                float s1 = ((kb + 16 + lx) <= qrow) ? sf[1][v] * scale_log2e : -1e30f;
                float mx   = half16_max(fmaxf(s0, s1));
                float mnew = fmaxf(rm[v], mx);
                alpha[v]   = exp2f(rm[v] - mnew);
                rm[v]      = mnew;
                float p0 = exp2f(s0 - mnew);
                float p1 = exp2f(s1 - mnew);
                sf[0][v] = p0;
                sf[1][v] = p1;
                rl[v] = rl[v] * alpha[v] + half16_sum(p0 + p1);
            }

            // ---- transpose P (C layout) -> A layout via per-wave LDS
            #pragma unroll
            for (int v = 0; v < 8; ++v) {
                pT[wv][v + hl * 8][lx]      = (__bf16)sf[0][v];
                pT[wv][v + hl * 8][16 + lx] = (__bf16)sf[1][v];
            }
            asm volatile("s_wait_dscnt 0" ::: "memory");  // per-wave LDS RAW fence

            const __bf16* pr = &pT[wv][lx][0];
            v8bf plo = *(const v8bf*)(pr + hl * 16);
            v8bf phi = *(const v8bf*)(pr + hl * 16 + 8);
            v16bf pa = __builtin_shufflevector(plo, phi,
                0, 1, 2, 3, 4, 5, 6, 7, 8, 9, 10, 11, 12, 13, 14, 15);

            // ---- acc = P V + diag(alpha) acc
            #pragma unroll
            for (int nt = 0; nt < 8; ++nt)
                #pragma unroll
                for (int v = 0; v < 8; ++v) acc[nt][v] *= alpha[v];

            #pragma unroll
            for (int nt = 0; nt < 8; ++nt) {
                v16bf vfrag = *(const v16bf*)&vT[nt * 16 + lx][hl * 16];
                acc[nt] = __builtin_amdgcn_wmma_f32_16x16x32_bf16(
                    false, pa, false, vfrag, (short)0, acc[nt], false, false);
            }
        }
        cur ^= 1;
    }

    // ---- normalize rows and store fp32 output
    float inv[8];
    #pragma unroll
    for (int v = 0; v < 8; ++v) inv[v] = 1.f / rl[v];
    #pragma unroll
    for (int v = 0; v < 8; ++v) {
        float* orow = O + (size_t)(qbase + v + hl * 8) * ROWSTRIDE + bh_off;
        #pragma unroll
        for (int nt = 0; nt < 8; ++nt)
            orow[nt * 16 + lx] = acc[nt][v] * inv[v];
    }
}

extern "C" void kernel_launch(void* const* d_in, const int* in_sizes, int n_in,
                              void* d_out, int out_size, void* d_ws, size_t ws_size,
                              hipStream_t stream) {
    const float* Q = (const float*)d_in[0];
    const float* K = (const float*)d_in[1];
    const float* V = (const float*)d_in[2];
    float* O = (float*)d_out;

    dim3 grid(S_LEN / 128, BATCH * HEADS);   // 16 x 64 blocks
    dim3 block(256);                          // 8 wave32
    fa_fwd_kernel<<<grid, block, 0, stream>>>(Q, K, V, O);
}
